// FVTA_9259949490210
// MI455X (gfx1250) — matrix operand order repaired and checked by reading, not compiled, not run on hardware
//
#include <hip/hip_runtime.h>

#define B_SZ 32
#define LQ 128
#define LK 4096
#define DK 256
#define DV 256
#define SPLIT 4
#define KPB (LK / SPLIT)            // keys per split-K block
#define NEGINF -1e30f

typedef __attribute__((ext_vector_type(16))) __bf16 v16bf;
typedef __attribute__((ext_vector_type(8)))  float  v8f;

union BF16x16 { v16bf v; uint4 q[2]; unsigned short u[16]; };

__device__ inline unsigned short f2bf(float f) {
    union { float f; unsigned u; } x; x.f = f;
    unsigned r = x.u + 0x7FFFu + ((x.u >> 16) & 1u);   // round-to-nearest-even
    return (unsigned short)(r >> 16);
}

// ---------- pass 1a: K fp32 [B][LK][DK] -> Kt bf16 [B][DK][LK] (transpose) ----------
__global__ __launch_bounds__(256)
void k_transpose_bf16(const float* __restrict__ K, unsigned short* __restrict__ Kt) {
    __shared__ float tile[32][33];
    const int b  = blockIdx.z;
    const int k0 = blockIdx.x * 32;
    const int d0 = blockIdx.y * 32;
    const int tx = threadIdx.x;        // 0..31
    const int ty = threadIdx.y;        // 0..7
#pragma unroll
    for (int i = 0; i < 4; ++i)
        tile[ty + 8 * i][tx] = K[((size_t)b * LK + k0 + ty + 8 * i) * DK + d0 + tx];
    __syncthreads();
#pragma unroll
    for (int i = 0; i < 4; ++i)
        Kt[((size_t)b * DK + d0 + ty + 8 * i) * LK + k0 + tx] = f2bf(tile[tx][ty + 8 * i]);
}

// ---------- pass 1b: V fp32 -> bf16 (same layout) ----------
__global__ __launch_bounds__(256)
void v_to_bf16(const float* __restrict__ V, unsigned short* __restrict__ Vb) {
    size_t i = ((size_t)blockIdx.x * 256 + threadIdx.x) * 8;
    float4 f0 = *(const float4*)(V + i);
    float4 f1 = *(const float4*)(V + i + 4);
    union { unsigned short u[8]; uint4 q; } o;
    o.u[0] = f2bf(f0.x); o.u[1] = f2bf(f0.y); o.u[2] = f2bf(f0.z); o.u[3] = f2bf(f0.w);
    o.u[4] = f2bf(f1.x); o.u[5] = f2bf(f1.y); o.u[6] = f2bf(f1.z); o.u[7] = f2bf(f1.w);
    *(uint4*)(Vb + i) = o.q;
}

// ---------- pass 2: split-K flash attention, bf16 WMMA ----------
__global__ __launch_bounds__(32)
void fvta_attn_split(const float* __restrict__ q,
                     const int*   __restrict__ qlen_,
                     const unsigned short* __restrict__ Kt,
                     const unsigned short* __restrict__ Vb,
                     const int*   __restrict__ mask,
                     float*       __restrict__ pstats,   // [nblk][16][{m,ssum}]
                     float*       __restrict__ pacc) {   // [nblk][16][DV]
    __shared__ unsigned short ldsQ[16][DK];   // bf16 Q tile (8 KB)
    __shared__ unsigned short ldsP[16][64];   // bf16 P tile (2 KB)

    const int lane = threadIdx.x;
    const int sp = blockIdx.x & (SPLIT - 1);
    const int qt = (blockIdx.x >> 2) & 7;
    const int b  = blockIdx.x >> 5;
    const int q0 = qt << 4;
    const int qlen = qlen_[b];
    const int col  = lane & 15;
    const bool hi  = lane >= 16;
    const int rbase = hi ? 8 : 0;             // C-layout: VGPR i -> row i (+8 for hi half)

    // stage Q tile to LDS as bf16
    const float* Qb = q + ((size_t)b * LQ + q0) * DK;
    for (int t = lane; t < 16 * DK; t += 32) {
        int r = t >> 8, d = t & (DK - 1);
        ldsQ[r][d] = f2bf(Qb[r * DK + d]);
    }
    __syncthreads();

    bool rowInv[8];
#pragma unroll
    for (int i = 0; i < 8; ++i) rowInv[i] = (q0 + rbase + i) >= qlen;

    float m[8], ssum[8];
#pragma unroll
    for (int i = 0; i < 8; ++i) { m[i] = NEGINF; ssum[i] = 0.f; }
    v8f accO[16];
#pragma unroll
    for (int j = 0; j < 16; ++j) { v8f z = {}; accO[j] = z; }

    const unsigned short* KtB = Kt + (size_t)b * DK * LK;
    const unsigned short* VbB = Vb + (size_t)b * LK * DV;
    const int* Mb = mask + (size_t)b * LK;

    const int kbeg = sp * KPB, kend = kbeg + KPB;
    for (int k0 = kbeg; k0 < kend; k0 += 64) {
        // ---- S = Q * K^T  (16 q-rows x 64 keys), A-frags as b128, K B-frags as 16-bit element loads ----
        v8f s[4];
#pragma unroll
        for (int t = 0; t < 4; ++t) { v8f z = {}; s[t] = z; }
#pragma unroll
        for (int c = 0; c < 8; ++c) {
            BF16x16 a;          // A-frag: halves contiguous at rbase / 16+rbase
            a.q[0] = *(const uint4*)&ldsQ[col][c * 32 + rbase];
            a.q[1] = *(const uint4*)&ldsQ[col][c * 32 + 16 + rbase];
            const unsigned short* kd = KtB + (size_t)(c * 32 + rbase) * LK + k0 + col;
#pragma unroll
            for (int t = 0; t < 4; ++t) {
                BF16x16 bk;     // B-frag: lane = key column, d halves at rbase / 16+rbase
#pragma unroll
                for (int s = 0; s < 16; ++s)
                    bk.u[s] = kd[(size_t)(s + (s >= 8 ? 8 : 0)) * LK + 16 * t];
                s[t] = __builtin_amdgcn_wmma_f32_16x16x32_bf16(false, a.v, false, bk.v,
                                                               (short)0, s[t], false, false);
            }
        }

        // ---- masking + online softmax over 64 keys ----
        bool mk[4];
#pragma unroll
        for (int t = 0; t < 4; ++t) mk[t] = Mb[k0 + 16 * t + col] != 0;

        float scl[8];
#pragma unroll
        for (int i = 0; i < 8; ++i) {
            const bool inv = rowInv[i];
            float mx = NEGINF;
#pragma unroll
            for (int t = 0; t < 4; ++t) {
                float a = (mk[t] | inv) ? NEGINF : s[t][i];
                mx = fmaxf(mx, a);
            }
#pragma unroll
            for (int off = 1; off < 16; off <<= 1)
                mx = fmaxf(mx, __shfl_xor(mx, off, 32));
            float mnew = fmaxf(m[i], mx);
            float sc   = __expf(m[i] - mnew);
            float ps   = 0.f;
#pragma unroll
            for (int t = 0; t < 4; ++t) {
                float p = (mk[t] | inv) ? 0.f : __expf(s[t][i] - mnew);
                ldsP[rbase + i][16 * t + col] = f2bf(p);
                ps += p;
            }
#pragma unroll
            for (int off = 1; off < 16; off <<= 1)
                ps += __shfl_xor(ps, off, 32);
            ssum[i] = ssum[i] * sc + ps;
            m[i]    = mnew;
            scl[i]  = sc;
        }
        v8f scv;
#pragma unroll
        for (int i = 0; i < 8; ++i) scv[i] = scl[i];
#pragma unroll
        for (int j = 0; j < 16; ++j) accO[j] *= scv;

        __syncthreads();

        // ---- O += P * V : two 16x32 A-frags, V B-frags as 16-bit element loads ----
        BF16x16 pA0, pA1;
        pA0.q[0] = *(const uint4*)&ldsP[col][rbase];
        pA0.q[1] = *(const uint4*)&ldsP[col][16 + rbase];
        pA1.q[0] = *(const uint4*)&ldsP[col][32 + rbase];
        pA1.q[1] = *(const uint4*)&ldsP[col][48 + rbase];

        const unsigned short* vr0 = VbB + (size_t)(k0 + rbase) * DV + col;
        const unsigned short* vr1 = vr0 + (size_t)32 * DV;
#pragma unroll
        for (int j = 0; j < 16; ++j) {
            BF16x16 b0, b1;
#pragma unroll
            for (int s = 0; s < 16; ++s) {
                const size_t o = (size_t)(s + (s >= 8 ? 8 : 0)) * DV + 16 * j;
                b0.u[s] = vr0[o];
                b1.u[s] = vr1[o];
            }
            accO[j] = __builtin_amdgcn_wmma_f32_16x16x32_bf16(false, pA0.v, false, b0.v,
                                                              (short)0, accO[j], false, false);
            accO[j] = __builtin_amdgcn_wmma_f32_16x16x32_bf16(false, pA1.v, false, b1.v,
                                                              (short)0, accO[j], false, false);
        }
        __syncthreads();
    }

    // ---- write split-K partials ----
    float* st = pstats + (size_t)blockIdx.x * 32;          // 16 rows * {m, ssum}
    float* pa = pacc   + (size_t)blockIdx.x * 16 * DV;
    if (col == 0) {
#pragma unroll
        for (int i = 0; i < 8; ++i) {
            st[(rbase + i) * 2 + 0] = m[i];
            st[(rbase + i) * 2 + 1] = ssum[i];
        }
    }
#pragma unroll
    for (int j = 0; j < 16; ++j)
#pragma unroll
        for (int i = 0; i < 8; ++i)
            pa[(size_t)(rbase + i) * DV + j * 16 + col] = accO[j][i];
}

// ---------- pass 3: merge split-K partials, mean over queries ----------
__global__ __launch_bounds__(32)
void fvta_combine(const float* __restrict__ pstats,
                  const float* __restrict__ pacc,
                  float*       __restrict__ out) {
    const int bq   = blockIdx.x;           // b*8 + qt
    const int b    = bq >> 3;
    const int lane = threadIdx.x;
    float oacc[8];
#pragma unroll
    for (int u = 0; u < 8; ++u) oacc[u] = 0.f;

    for (int r = 0; r < 16; ++r) {
        float M = NEGINF;
#pragma unroll
        for (int sp = 0; sp < SPLIT; ++sp)
            M = fmaxf(M, pstats[(size_t)(bq * SPLIT + sp) * 32 + r * 2]);
        float w[SPLIT];
        float tot = 0.f;
#pragma unroll
        for (int sp = 0; sp < SPLIT; ++sp) {
            float mm = pstats[(size_t)(bq * SPLIT + sp) * 32 + r * 2];
            float ss = pstats[(size_t)(bq * SPLIT + sp) * 32 + r * 2 + 1];
            w[sp] = __expf(mm - M);
            tot += w[sp] * ss;
        }
        float invt = tot > 0.f ? 1.f / tot : 0.f;
#pragma unroll
        for (int u = 0; u < 8; ++u) {
            int dv = lane + 32 * u;
            float v = 0.f;
#pragma unroll
            for (int sp = 0; sp < SPLIT; ++sp)
                v += w[sp] * pacc[(size_t)(bq * SPLIT + sp) * 16 * DV + (size_t)r * DV + dv];
            oacc[u] += v * invt;
        }
    }
#pragma unroll
    for (int u = 0; u < 8; ++u)
        atomicAdd(&out[(size_t)b * DV + lane + 32 * u], oacc[u] * (1.0f / LQ));
}

// ---------- fallback: fused single-pass kernel (no workspace) ----------
__global__ __launch_bounds__(32)
void fvta_attn_fused(const float* __restrict__ q,
                     const int*   __restrict__ qlen_,
                     const float* __restrict__ kmat,
                     const float* __restrict__ vmat,
                     const int*   __restrict__ mask,
                     float*       __restrict__ out) {
    __shared__ unsigned short ldsQ[16][DK];
    __shared__ unsigned short ldsP[16][32];

    const int lane = threadIdx.x;
    const int b    = blockIdx.x >> 3;
    const int q0   = (blockIdx.x & 7) << 4;
    const int qlen = qlen_[b];
    const int col  = lane & 15;
    const bool hi  = lane >= 16;
    const int rbase = hi ? 8 : 0;

    const float* Qb = q + ((size_t)b * LQ + q0) * DK;
    for (int t = lane; t < 16 * DK; t += 32) {
        int r = t >> 8, d = t & (DK - 1);
        ldsQ[r][d] = f2bf(Qb[r * DK + d]);
    }
    __syncthreads();

    bool rowInv[8];
#pragma unroll
    for (int i = 0; i < 8; ++i) rowInv[i] = (q0 + rbase + i) >= qlen;
    float m[8], ssum[8];
#pragma unroll
    for (int i = 0; i < 8; ++i) { m[i] = NEGINF; ssum[i] = 0.f; }
    v8f accO[16];
#pragma unroll
    for (int j = 0; j < 16; ++j) { v8f z = {}; accO[j] = z; }

    int kofs[16];
#pragma unroll
    for (int vv = 0; vv < 8; ++vv)
#pragma unroll
        for (int h = 0; h < 2; ++h)
            kofs[2 * vv + h] = ((vv >= 4) ? 16 : 0) + rbase + ((vv & 3) << 1) + h;

    const float* Kb = kmat + (size_t)b * LK * DK;
    const float* Vb = vmat + (size_t)b * LK * DV;
    const int*   Mb = mask + (size_t)b * LK;

    for (int k0 = 0; k0 < LK; k0 += 32) {
        v8f s0 = {}, s1 = {};
#pragma unroll
        for (int c = 0; c < 8; ++c) {
            BF16x16 a;
#pragma unroll
            for (int j = 0; j < 16; ++j) a.u[j] = ldsQ[col][c * 32 + kofs[j]];
            BF16x16 bk0, bk1;
            const int d = c * 32 + lane;
#pragma unroll
            for (int n = 0; n < 16; ++n) {
                bk0.u[n] = f2bf(Kb[(size_t)(k0 + col)      * DK + c * 32 + kofs[n]]);
                bk1.u[n] = f2bf(Kb[(size_t)(k0 + 16 + col) * DK + c * 32 + kofs[n]]);
            }
            s0 = __builtin_amdgcn_wmma_f32_16x16x32_bf16(false, a.v, false, bk0.v, (short)0, s0, false, false);
            s1 = __builtin_amdgcn_wmma_f32_16x16x32_bf16(false, a.v, false, bk1.v, (short)0, s1, false, false);
        }
        const bool mk0 = Mb[k0 + col] != 0;
        const bool mk1 = Mb[k0 + 16 + col] != 0;
        float scl[8];
#pragma unroll
        for (int i = 0; i < 8; ++i) {
            const bool z0 = mk0 | rowInv[i], z1 = mk1 | rowInv[i];
            float a0 = z0 ? NEGINF : s0[i];
            float a1 = z1 ? NEGINF : s1[i];
            float mx = fmaxf(a0, a1);
#pragma unroll
            for (int off = 1; off < 16; off <<= 1) mx = fmaxf(mx, __shfl_xor(mx, off, 32));
            float mnew = fmaxf(m[i], mx);
            float sc = __expf(m[i] - mnew);
            float p0 = z0 ? 0.f : __expf(s0[i] - mnew);
            float p1 = z1 ? 0.f : __expf(s1[i] - mnew);
            float ps = p0 + p1;
#pragma unroll
            for (int off = 1; off < 16; off <<= 1) ps += __shfl_xor(ps, off, 32);
            ssum[i] = ssum[i] * sc + ps; m[i] = mnew; scl[i] = sc;
            ldsP[rbase + i][col] = f2bf(p0);
            ldsP[rbase + i][16 + col] = f2bf(p1);
        }
        v8f scv;
#pragma unroll
        for (int i = 0; i < 8; ++i) scv[i] = scl[i];
#pragma unroll
        for (int j = 0; j < 16; ++j) accO[j] *= scv;
        __syncthreads();
        BF16x16 pA;
#pragma unroll
        for (int j = 0; j < 16; ++j) pA.u[j] = ldsP[col][kofs[j]];
        const float* Vrow = Vb + (size_t)(k0 + lane) * DV;
#pragma unroll
        for (int j = 0; j < 16; ++j) {
            BF16x16 bv;
#pragma unroll
            for (int t = 0; t < 16; ++t)
                bv.u[t] = f2bf(Vb[(size_t)(k0 + kofs[t]) * DV + j * 16 + col]);
            accO[j] = __builtin_amdgcn_wmma_f32_16x16x32_bf16(false, pA.v, false, bv.v, (short)0, accO[j], false, false);
        }
        __syncthreads();
    }
    float invs[8];
#pragma unroll
    for (int i = 0; i < 8; ++i) invs[i] = ssum[i] > 0.f ? 1.f / ssum[i] : 0.f;
#pragma unroll
    for (int j = 0; j < 16; ++j) {
        float part = 0.f;
#pragma unroll
        for (int i = 0; i < 8; ++i) part += accO[j][i] * invs[i];
        atomicAdd(&out[(size_t)b * DV + j * 16 + col], part * (1.0f / LQ));
    }
}

extern "C" void kernel_launch(void* const* d_in, const int* in_sizes, int n_in,
                              void* d_out, int out_size, void* d_ws, size_t ws_size,
                              hipStream_t stream) {
    const float* q    = (const float*)d_in[0];
    const int*   qlen = (const int*)  d_in[1];
    const float* k    = (const float*)d_in[2];
    const float* v    = (const float*)d_in[3];
    const int*   mask = (const int*)  d_in[4];
    float* out = (float*)d_out;

    const size_t ktBytes = (size_t)B_SZ * DK * LK * 2;                 // 64 MB
    const size_t vbBytes = (size_t)B_SZ * LK * DV * 2;                 // 64 MB
    const size_t nblk    = (size_t)B_SZ * 8 * SPLIT;                   // 1024
    const size_t stBytes = nblk * 32 * sizeof(float);
    const size_t paBytes = nblk * 16 * DV * sizeof(float);
    const size_t need    = ktBytes + vbBytes + stBytes + paBytes;

    hipMemsetAsync(out, 0, (size_t)out_size * sizeof(float), stream);

    if (ws_size >= need) {
        unsigned short* Kt = (unsigned short*)d_ws;
        unsigned short* Vb = (unsigned short*)((char*)d_ws + ktBytes);
        float* pstats = (float*)((char*)d_ws + ktBytes + vbBytes);
        float* pacc   = (float*)((char*)d_ws + ktBytes + vbBytes + stBytes);

        k_transpose_bf16<<<dim3(LK / 32, DK / 32, B_SZ), dim3(32, 8), 0, stream>>>(k, Kt);
        v_to_bf16<<<dim3((unsigned)((size_t)B_SZ * LK * DV / 8 / 256)), dim3(256), 0, stream>>>(v, Vb);
        fvta_attn_split<<<dim3((unsigned)nblk), dim3(32), 0, stream>>>(
            q, qlen, Kt, Vb, mask, pstats, pacc);
        fvta_combine<<<dim3(B_SZ * 8), dim3(32), 0, stream>>>(pstats, pacc, out);
    } else {
        fvta_attn_fused<<<dim3(B_SZ * (LQ / 16)), dim3(32), 0, stream>>>(
            q, qlen, k, v, mask, out);
    }
}
